// ImprovedEdgeGNN_8710193676615
// MI455X (gfx1250) — matrix operand order repaired and checked by reading, not compiled, not run on hardware
//
#include <hip/hip_runtime.h>
#include <hip/hip_bf16.h>

// ---------------------------------------------------------------------------
// ImprovedEdgeGNN for MI455X (gfx1250, wave32, WMMA)
//   K1: fused GEMM  out[N,192] = x[N,256] @ [we_src|we_dst|wc]  (bf16 WMMA)
//       cols 0..31 -> ps, 32..63 -> pd, 64..191 -> h = relu(.+bc)
//       B fragments staged in LDS (96 KB), software-pipelined ds loads
//   K2: warp-per-edge gate + gated scatter-add (agg, deg)
//   K3: softmax(deg) reductions + attention pooling -> g[128]
//   K4: classifier with closed-form spectral norm of w2 (64x2)
// ---------------------------------------------------------------------------

typedef __attribute__((ext_vector_type(16))) __bf16 v16bf;
typedef __attribute__((ext_vector_type(8)))  float  v8f;
typedef __attribute__((ext_vector_type(4)))  float  v4f;

#define FDIM 256
#define HDIM 128
#define EDIM 32
#define NTILES 12         // 192 / 16 fused output column tiles
#define KSTEPS 8          // 256 / 32
#define NPART  512

__device__ __forceinline__ __bf16 f2bf(float f) {
    unsigned u = __builtin_bit_cast(unsigned, f);
    u += 0x7FFFu + ((u >> 16) & 1u);          // round-to-nearest-even
    unsigned short hs = (unsigned short)(u >> 16);
    __bf16 r;
    __builtin_memcpy(&r, &hs, 2);
    return r;
}

// ---------------- K1: fused node GEMM (bf16 WMMA, weights staged in LDS) ----
__global__ __launch_bounds__(256)
void gemm_fused_kernel(const float* __restrict__ x,
                       const float* __restrict__ we_src,
                       const float* __restrict__ we_dst,
                       const float* __restrict__ wc,
                       const float* __restrict__ bc,
                       float* __restrict__ ps,
                       float* __restrict__ pd,
                       float* __restrict__ h,
                       int n)
{
    // Packed bf16 weight fragments: fragment id f = ks*NTILES + nt so the
    // inner nt-loop walks 12 contiguous KB (fits DS 16-bit immediate offset).
    // Each fragment: 32 lanes x 16 bf16 = 1 KB -> 96 KB LDS total.
    __shared__ unsigned short wlds[NTILES * KSTEPS * 512];

    const int tid  = threadIdx.x;
    const int lane = tid & 31;
    const int wave = tid >> 5;

    // --- cooperative pack: W[256,192] -> LDS bf16 in exact B-fragment order
    for (int gi = tid; gi < NTILES * KSTEPS * 512; gi += 256) {
        int f    = gi >> 9;          // fragment id = ks*NTILES + nt
        int rem  = gi & 511;
        int fl   = rem >> 4;         // lane within fragment
        int i    = rem & 15;         // element within lane vector
        int ks   = f / NTILES;
        int nt   = f - ks * NTILES;
        int kbase = (fl < 16) ? 0 : 8;
        int r = ks * 32 + kbase + (i & 7) + ((i >> 3) << 4);  // K index (0..255)
        int c = nt * 16 + (fl & 15);         // fused column (0..191)
        float v;
        if (c < 32)       v = we_src[r * EDIM + c];
        else if (c < 64)  v = we_dst[r * EDIM + (c - 32)];
        else              v = wc[r * HDIM + (c - 64)];
        __bf16 bv = f2bf(v);
        unsigned short us;
        __builtin_memcpy(&us, &bv, 2);
        wlds[gi] = us;
    }
    __syncthreads();

    // --- each wave computes a 16-row strip across all 12 column tiles
    const int rowbase = blockIdx.x * 128 + wave * 16;
    int arow = rowbase + (lane & 15);
    if (arow > n - 1) arow = n - 1;           // clamp; stores are guarded
    const float* xr = x + (size_t)arow * FDIM;
    const int klo = (lane < 16) ? 0 : 8;

    v8f acc[NTILES] = {};

    #pragma unroll
    for (int ks = 0; ks < KSTEPS; ++ks) {
        const float* xk = xr + ks * 32;
        // A fragment (16-bit A 16x32 layout): lanes<16 K={0..7,16..23},
        // lanes>=16 K={8..15,24..31}; two contiguous float4 pairs per lane.
        v4f p0 = *(const v4f*)(xk + klo);
        v4f p1 = *(const v4f*)(xk + klo + 4);
        v4f p2 = *(const v4f*)(xk + 16 + klo);
        v4f p3 = *(const v4f*)(xk + 16 + klo + 4);
        v16bf a;
        #pragma unroll
        for (int i = 0; i < 4; ++i) {
            a[i]      = f2bf(p0[i]);
            a[4 + i]  = f2bf(p1[i]);
            a[8 + i]  = f2bf(p2[i]);
            a[12 + i] = f2bf(p3[i]);
        }

        // Software-pipelined B fragments: load nt+1 while WMMA consumes nt,
        // so the compiler can wait dscnt<=2 instead of 0 per WMMA.
        const unsigned short* wp = &wlds[(ks * NTILES) * 512 + lane * 16];
        v16bf bcur = *reinterpret_cast<const v16bf*>(wp);
        #pragma unroll
        for (int nt = 0; nt < NTILES; ++nt) {
            v16bf bnext;
            if (nt + 1 < NTILES)
                bnext = *reinterpret_cast<const v16bf*>(wp + (nt + 1) * 512);
            else
                bnext = bcur;                 // dead on last iteration
            acc[nt] = __builtin_amdgcn_wmma_f32_16x16x32_bf16(
                false, a, false, bcur, (short)0, acc[nt], false, false);
            bcur = bnext;
        }
    }

    // --- epilogue: C/D layout -> scatter into ps / pd / h
    const int mofs = (lane >> 4) << 3;        // 0 or 8
    const int ncol = lane & 15;
    #pragma unroll
    for (int nt = 0; nt < NTILES; ++nt) {
        const int col = nt * 16 + ncol;
        #pragma unroll
        for (int i = 0; i < 8; ++i) {
            const int row = rowbase + mofs + i;
            if (row < n) {
                float v = acc[nt][i];
                if (col < 32) {
                    ps[row * EDIM + col] = v;
                } else if (col < 64) {
                    pd[row * EDIM + (col - 32)] = v;
                } else {
                    int hc = col - 64;
                    float r = v + bc[hc];
                    h[row * HDIM + hc] = r > 0.f ? r : 0.f;
                }
            }
        }
    }
}

// ---------------- zero scratch ---------------------------------------------
__global__ void zero_kernel(float* __restrict__ p, long long n)
{
    long long i = (long long)blockIdx.x * blockDim.x + threadIdx.x;
    long long stride = (long long)gridDim.x * blockDim.x;
    for (; i < n; i += stride) p[i] = 0.f;
}

// ---------------- K2: warp-per-edge gate + gated message scatter -----------
__global__ __launch_bounds__(256)
void edge_kernel(const int* __restrict__ ei,        // [2,E] src then dst
                 const float* __restrict__ ps,
                 const float* __restrict__ pd,
                 const float* __restrict__ be,
                 const float* __restrict__ we2,
                 const float* __restrict__ be2,
                 const float* __restrict__ h,
                 float* __restrict__ agg,
                 float* __restrict__ deg,
                 int nE)
{
    const int lane = threadIdx.x & 31;
    long long warp = (long long)blockIdx.x * (blockDim.x >> 5) + (threadIdx.x >> 5);
    const long long nw = (long long)gridDim.x * (blockDim.x >> 5);
    const float bev  = be[lane];
    const float w2v  = we2[lane];
    const float be2v = be2[0];

    for (long long e = warp; e < nE; e += nw) {
        const int s = ei[e];
        const int d = ei[nE + e];
        float v = ps[s * EDIM + lane] + pd[d * EDIM + lane] + bev;
        v = fmaxf(v, 0.f);
        float p = v * w2v;
        p += __shfl_xor(p, 16);
        p += __shfl_xor(p, 8);
        p += __shfl_xor(p, 4);
        p += __shfl_xor(p, 2);
        p += __shfl_xor(p, 1);
        const float logit = p + be2v;
        float gate = 1.f / (1.f + expf(-logit));
        gate = gate * 1.2f - 0.1f;                 // (ZETA-GAMMA)=1.2, +GAMMA
        gate = fminf(fmaxf(gate, 0.f), 1.f);
        if (gate > 0.f) {                          // adding zero is a no-op
            const float* hr = h + (size_t)s * HDIM;
            float*       ar = agg + (size_t)d * HDIM;
            atomicAdd(&ar[lane],      gate * hr[lane]);
            atomicAdd(&ar[lane + 32], gate * hr[lane + 32]);
            atomicAdd(&ar[lane + 64], gate * hr[lane + 64]);
            atomicAdd(&ar[lane + 96], gate * hr[lane + 96]);
            if (lane == 0) atomicAdd(&deg[d], gate);
        }
    }
}

// ---------------- K3: softmax(deg) reductions + pooling --------------------
__global__ void redmax_kernel(const float* __restrict__ deg, float* __restrict__ pmax, int n)
{
    __shared__ float sh[256];
    float m = -1e30f;
    for (int i = blockIdx.x * blockDim.x + threadIdx.x; i < n; i += gridDim.x * blockDim.x)
        m = fmaxf(m, deg[i]);
    sh[threadIdx.x] = m; __syncthreads();
    for (int s = 128; s > 0; s >>= 1) {
        if (threadIdx.x < s) sh[threadIdx.x] = fmaxf(sh[threadIdx.x], sh[threadIdx.x + s]);
        __syncthreads();
    }
    if (threadIdx.x == 0) pmax[blockIdx.x] = sh[0];
}

__global__ void finalmax_kernel(const float* __restrict__ pmax, float* __restrict__ scal, int np)
{
    __shared__ float sh[256];
    float m = -1e30f;
    for (int i = threadIdx.x; i < np; i += blockDim.x) m = fmaxf(m, pmax[i]);
    sh[threadIdx.x] = m; __syncthreads();
    for (int s = 128; s > 0; s >>= 1) {
        if (threadIdx.x < s) sh[threadIdx.x] = fmaxf(sh[threadIdx.x], sh[threadIdx.x + s]);
        __syncthreads();
    }
    if (threadIdx.x == 0) scal[0] = sh[0];
}

__global__ void redsum_kernel(const float* __restrict__ deg, const float* __restrict__ scal,
                              float* __restrict__ psum, int n)
{
    __shared__ float sh[256];
    const float mx = scal[0];
    float s = 0.f;
    for (int i = blockIdx.x * blockDim.x + threadIdx.x; i < n; i += gridDim.x * blockDim.x)
        s += expf(deg[i] - mx);
    sh[threadIdx.x] = s; __syncthreads();
    for (int t = 128; t > 0; t >>= 1) {
        if (threadIdx.x < t) sh[threadIdx.x] += sh[threadIdx.x + t];
        __syncthreads();
    }
    if (threadIdx.x == 0) psum[blockIdx.x] = sh[0];
}

__global__ void finalsum_kernel(const float* __restrict__ psum, float* __restrict__ scal, int np)
{
    __shared__ float sh[256];
    float s = 0.f;
    for (int i = threadIdx.x; i < np; i += blockDim.x) s += psum[i];
    sh[threadIdx.x] = s; __syncthreads();
    for (int t = 128; t > 0; t >>= 1) {
        if (threadIdx.x < t) sh[threadIdx.x] += sh[threadIdx.x + t];
        __syncthreads();
    }
    if (threadIdx.x == 0) scal[1] = sh[0];
}

__global__ __launch_bounds__(256)
void pool_kernel(const float* __restrict__ h, const float* __restrict__ agg,
                 const float* __restrict__ deg, const float* __restrict__ scal,
                 float* __restrict__ g, int n)
{
    const int lane = threadIdx.x & 31;
    long long warp = (long long)blockIdx.x * (blockDim.x >> 5) + (threadIdx.x >> 5);
    const long long nw = (long long)gridDim.x * (blockDim.x >> 5);
    const float mx = scal[0];
    const float invZ = 1.f / scal[1];
    float a0 = 0.f, a1 = 0.f, a2 = 0.f, a3 = 0.f;
    for (long long i = warp; i < n; i += nw) {
        const float dg  = deg[i];
        const float w   = expf(dg - mx) * invZ;
        const float inv = 1.f / (dg + 1e-6f);
        const float* hr = h   + (size_t)i * HDIM;
        const float* ar = agg + (size_t)i * HDIM;
        a0 += w * (hr[lane]      + ar[lane]      * inv);
        a1 += w * (hr[lane + 32] + ar[lane + 32] * inv);
        a2 += w * (hr[lane + 64] + ar[lane + 64] * inv);
        a3 += w * (hr[lane + 96] + ar[lane + 96] * inv);
    }
    atomicAdd(&g[lane],      a0);
    atomicAdd(&g[lane + 32], a1);
    atomicAdd(&g[lane + 64], a2);
    atomicAdd(&g[lane + 96], a3);
}

// ---------------- K4: classifier + analytic spectral norm ------------------
__global__ void cls_kernel(const float* __restrict__ g,
                           const float* __restrict__ w1, const float* __restrict__ b1,
                           const float* __restrict__ ln_g, const float* __restrict__ ln_b,
                           const float* __restrict__ w2, const float* __restrict__ b2,
                           float* __restrict__ out)
{
    __shared__ float sh[64];
    __shared__ float st[8];
    const int j = threadIdx.x;   // 0..63
    float dot = 0.f;
    for (int k = 0; k < HDIM; ++k) dot += g[k] * w1[k * 64 + j];
    const float z = fmaxf(dot + b1[j], 0.f);

    sh[j] = z; __syncthreads();
    if (j == 0) { float s = 0.f; for (int t = 0; t < 64; ++t) s += sh[t]; st[0] = s * (1.f / 64.f); }
    __syncthreads();
    const float mn = st[0];
    const float d  = z - mn;
    sh[j] = d * d; __syncthreads();
    if (j == 0) { float s = 0.f; for (int t = 0; t < 64; ++t) s += sh[t]; st[1] = s * (1.f / 64.f); }
    __syncthreads();
    const float zn = d / sqrtf(st[1] + 1e-5f) * ln_g[j] + ln_b[j];

    const float wA = w2[j * 2 + 0];
    const float wB = w2[j * 2 + 1];
    sh[j] = wA * wA; __syncthreads();
    if (j == 0) { float s = 0.f; for (int t = 0; t < 64; ++t) s += sh[t]; st[2] = s; }
    __syncthreads();
    sh[j] = wA * wB; __syncthreads();
    if (j == 0) { float s = 0.f; for (int t = 0; t < 64; ++t) s += sh[t]; st[3] = s; }
    __syncthreads();
    sh[j] = wB * wB; __syncthreads();
    if (j == 0) { float s = 0.f; for (int t = 0; t < 64; ++t) s += sh[t]; st[4] = s; }
    __syncthreads();
    sh[j] = zn * wA; __syncthreads();
    if (j == 0) { float s = 0.f; for (int t = 0; t < 64; ++t) s += sh[t]; st[5] = s; }
    __syncthreads();
    sh[j] = zn * wB; __syncthreads();
    if (j == 0) { float s = 0.f; for (int t = 0; t < 64; ++t) s += sh[t]; st[6] = s; }
    __syncthreads();

    if (j == 0) {
        // spectral norm of 64x2 w2: sqrt(max eigenvalue of 2x2 Gram)
        const float a = st[2], b = st[3], c = st[4];
        const float disc = sqrtf((a - c) * (a - c) + 4.f * b * b);
        const float sigma = sqrtf(0.5f * ((a + c) + disc));
        out[0] = st[5] / sigma + b2[0];
        out[1] = st[6] / sigma + b2[1];
    }
}

// ---------------------------------------------------------------------------
extern "C" void kernel_launch(void* const* d_in, const int* in_sizes, int n_in,
                              void* d_out, int out_size, void* d_ws, size_t ws_size,
                              hipStream_t stream)
{
    (void)n_in; (void)out_size; (void)ws_size;
    const float* x      = (const float*)d_in[0];
    const int*   ei     = (const int*)d_in[1];
    const float* we_src = (const float*)d_in[2];
    const float* we_dst = (const float*)d_in[3];
    const float* be     = (const float*)d_in[4];
    const float* we2    = (const float*)d_in[5];
    const float* be2    = (const float*)d_in[6];
    const float* wc     = (const float*)d_in[7];
    const float* bc     = (const float*)d_in[8];
    const float* w1     = (const float*)d_in[9];
    const float* b1     = (const float*)d_in[10];
    const float* ln_g   = (const float*)d_in[11];
    const float* ln_b   = (const float*)d_in[12];
    const float* w2     = (const float*)d_in[13];
    const float* b2     = (const float*)d_in[14];
    float* out = (float*)d_out;

    const int N = in_sizes[0] / FDIM;
    const int E = in_sizes[1] / 2;

    // workspace layout (floats)
    float* ws   = (float*)d_ws;
    float* ps   = ws;                        // N*32
    float* pd   = ps  + (size_t)N * EDIM;    // N*32
    float* h    = pd  + (size_t)N * EDIM;    // N*128
    float* agg  = h   + (size_t)N * HDIM;    // N*128   (zeroed)
    float* deg  = agg + (size_t)N * HDIM;    // N       (zeroed)
    float* g    = deg + N;                   // 128     (zeroed)
    float* pmax = g + HDIM;                  // NPART
    float* psum = pmax + NPART;              // NPART
    float* scal = psum + NPART;              // [0]=max  [1]=sumexp

    const long long nzero = (long long)N * HDIM + N + HDIM;  // agg,deg,g contiguous
    zero_kernel<<<1024, 256, 0, stream>>>(agg, nzero);

    gemm_fused_kernel<<<(N + 127) / 128, 256, 0, stream>>>(
        x, we_src, we_dst, wc, bc, ps, pd, h, N);

    edge_kernel<<<2048, 256, 0, stream>>>(ei, ps, pd, be, we2, be2, h, agg, deg, E);

    redmax_kernel<<<NPART, 256, 0, stream>>>(deg, pmax, N);
    finalmax_kernel<<<1, 256, 0, stream>>>(pmax, scal, NPART);
    redsum_kernel<<<NPART, 256, 0, stream>>>(deg, scal, psum, N);
    finalsum_kernel<<<1, 256, 0, stream>>>(psum, scal, NPART);

    pool_kernel<<<512, 256, 0, stream>>>(h, agg, deg, scal, g, N);

    cls_kernel<<<1, 64, 0, stream>>>(g, w1, b1, ln_g, ln_b, w2, b2, out);
}
